// CTViT_9740985827394
// MI455X (gfx1250) — matrix operand order repaired and checked by reading, not compiled, not run on hardware
//
#include <hip/hip_runtime.h>
#include <hip/hip_bf16.h>

// ---------------- model constants ----------------
#define DIMV   512
#define HEADS  8
#define DH     64
#define TFR    17          // 1 + (33-1)/2 frames of tokens
#define BSZ    2
#define NTOK   (BSZ*TFR*16*16)    // 8704 tokens total
#define XSZ    (NTOK*DIMV)        // 4,456,448 floats
#define FFI    1365
#define FFIP   1376               // FFI padded to mult of 32 (K of fw2)
#define FF2N   2730
#define FF2NP  2736               // fw1 N padded to mult of 16
#define QKVPAD (16384*DIMV)       // padded swizzled qkv buffer (512 seq * 32 pad tokens)

// ---------------- WMMA plumbing ----------------
typedef _Float16 half16 __attribute__((ext_vector_type(16)));
typedef float    float8 __attribute__((ext_vector_type(8)));
typedef int      i32x4  __attribute__((ext_vector_type(4)));

union AF { half16 v; _Float16 e[16]; };
union CF { float8 v; float    e[8];  };

static __device__ __forceinline__ float8 wmma_f16(half16 a, half16 b, float8 c) {
  // D(16x16,f32) = A(16x32,f16) x B(32x16,f16) + C
  return __builtin_amdgcn_wmma_f32_16x16x32_f16(false, a, false, b, (short)0, c, false, false);
}

// Fragment index maps (ISA 7.12.2):
//  A (16x32, f16): element (m,k): lane = (m&15) + 16*((k>>3)&1), e = ((k>>4)<<3) + (k&7)
//  B (32x16, f16): element (k,n): lane = 16*((k>>3)&1) + (n&15), e = ((k>>4)<<3) + (k&7)
//  C/D (16x16,f32): element (m,n): lane = 16*((m>>3)&1) + (n&15), vgpr r = m&7

#if __has_builtin(__builtin_amdgcn_global_load_async_to_lds_b128)
#define USE_ASYNC_COPY 1
typedef __attribute__((address_space(1))) i32x4* as1_i32x4p;
typedef __attribute__((address_space(3))) i32x4* as3_i32x4p;
#else
#define USE_ASYNC_COPY 0
#endif

// ---------------- generic GEMM: C[M,N] = A[M,K]fp32 * Bsw[K,N]f16 (+bias)(+res)(+leaky)
// Bsw is pre-swizzled: tile (kt=k/32, nt=n/16) of 512 halves at
//   ((kt*(N/16)+nt)*32 + lane)*16 + e   (B-fragment order)
#define BM 128
#define BN 128
#define KT 32

__global__ __launch_bounds__(256)
void k_gemm(const float* __restrict__ A, const _Float16* __restrict__ Bsw,
            const float* __restrict__ bias, const float* __restrict__ res,
            float* __restrict__ C, int M, int N, int K, int act)
{
  __shared__ _Float16 As[BM * KT];      // 4096 halves, A-fragment blocked
  __shared__ _Float16 Bs[KT * BN];      // 4096 halves, B-fragment blocked
  const int tid  = threadIdx.x;
  const int lane = tid & 31;
  const int wid  = tid >> 5;
  const int wy = wid & 3;               // 32-row block of wave
  const int wx = wid >> 2;              // 64-col block of wave
  const int hi  = (lane >> 4) & 1;
  const int l15 = lane & 15;
  const int m0 = blockIdx.y * BM;
  const int n0 = blockIdx.x * BN;
  const int ntiles = N >> 4;
  int validTiles = ntiles - (n0 >> 4);
  if (validTiles > 8) validTiles = 8;
  const int validHalves = validTiles * 512;

  CF acc[2][4];
  #pragma unroll
  for (int s = 0; s < 2; ++s)
    #pragma unroll
    for (int t = 0; t < 4; ++t)
      #pragma unroll
      for (int r = 0; r < 8; ++r) acc[s][t].e[r] = 0.f;

  const int arow  = tid >> 1;           // 0..127
  const int acol0 = (tid & 1) * 16;
  const int mt = arow >> 4;
  const int mr = arow & 15;

  for (int kb = 0; kb < K; kb += KT) {
    // ---- B tile: contiguous copy of pre-swizzled fragments ----
    const _Float16* bsrc = Bsw + ((size_t)(kb >> 5) * ntiles + (n0 >> 4)) * 512;
#if USE_ASYNC_COPY
    #pragma unroll
    for (int c = 0; c < 2; ++c) {
      int ch = tid + c * 256;           // 512 chunks of 8 halves (16B)
      if (ch * 8 < validHalves) {
        __builtin_amdgcn_global_load_async_to_lds_b128(
            (as1_i32x4p)(bsrc + ch * 8),
            (as3_i32x4p)&Bs[ch * 8], 0, 0);
      }
    }
#else
    #pragma unroll
    for (int c = 0; c < 2; ++c) {
      int ch = tid + c * 256;
      if (ch * 8 < validHalves)
        *(uint4*)&Bs[ch * 8] = *(const uint4*)(bsrc + ch * 8);
    }
#endif
    // ---- A tile: fp32 global -> f16 LDS, written in A-fragment order ----
    {
      const int gm = m0 + arow;
      const bool ok = gm < M;
      const float* ap = A + (size_t)gm * K + kb + acol0;
      if (ok && kb + KT < K) __builtin_prefetch(ap + KT, 0, 1);  // global_prefetch_b8
      #pragma unroll
      for (int i = 0; i < 4; ++i) {
        float4 vv = ok ? *(const float4*)(ap + i * 4) : make_float4(0.f, 0.f, 0.f, 0.f);
        int k0 = acol0 + i * 4;                       // 4 consecutive k, same hi bit
        int lA = mr + (((k0 >> 3) & 1) << 4);
        int e0 = ((k0 >> 4) << 3) + (k0 & 7);         // 4 consecutive e values
        union { _Float16 h[4]; uint2 u; } pk;
        pk.h[0] = (_Float16)vv.x; pk.h[1] = (_Float16)vv.y;
        pk.h[2] = (_Float16)vv.z; pk.h[3] = (_Float16)vv.w;
        *(uint2*)&As[(mt * 32 + lA) * 16 + e0] = pk.u;
      }
    }
#if USE_ASYNC_COPY
#if __has_builtin(__builtin_amdgcn_s_wait_asynccnt)
    __builtin_amdgcn_s_wait_asynccnt(0);
#else
    asm volatile("s_wait_asynccnt 0" ::: "memory");
#endif
#endif
    __syncthreads();

    // ---- fragments: two ds_load_b128 each ----
    AF a[2], b[4];
    #pragma unroll
    for (int s = 0; s < 2; ++s) {
      const _Float16* p = &As[((wy * 2 + s) * 32 + lane) * 16];
      *(uint4*)&a[s].e[0] = *(const uint4*)p;
      *(uint4*)&a[s].e[8] = *(const uint4*)(p + 8);
    }
    #pragma unroll
    for (int t = 0; t < 4; ++t) {
      const _Float16* p = &Bs[((wx * 4 + t) * 32 + lane) * 16];
      *(uint4*)&b[t].e[0] = *(const uint4*)p;
      *(uint4*)&b[t].e[8] = *(const uint4*)(p + 8);
    }
    #pragma unroll
    for (int s = 0; s < 2; ++s)
      #pragma unroll
      for (int t = 0; t < 4; ++t)
        acc[s][t].v = wmma_f16(a[s].v, b[t].v, acc[s][t].v);
    __syncthreads();
  }

  // ---- epilogue ----
  #pragma unroll
  for (int s = 0; s < 2; ++s) {
    #pragma unroll
    for (int t = 0; t < 4; ++t) {
      int n = n0 + wx * 64 + t * 16 + l15;
      #pragma unroll
      for (int r = 0; r < 8; ++r) {
        int m = m0 + wy * 32 + s * 16 + r + hi * 8;
        if (m < M && n < N) {
          float v = acc[s][t].e[r];
          if (bias) v += bias[n];
          if (res)  v += res[(size_t)m * N + n];
          if (act == 1) v = (v < 0.f) ? 0.1f * v : v;   // leaky relu 0.1
          C[(size_t)m * N + n] = v;
        }
      }
    }
  }
}

// ---------------- fused cosine-sim attention ----------------
// qsw: A-frag blocked (per seq,head: [qt][c]{512}), ksw: B-frag blocked ([jt][c]{512}),
// vsw: B-frag blocked over (key-chunk, dim-tile): ([kk][dt]{512}).
__global__ __launch_bounds__(32)
void k_attn(const _Float16* __restrict__ qsw, const _Float16* __restrict__ ksw,
            const _Float16* __restrict__ vsw, const float* __restrict__ btab,
            float* __restrict__ out, int n, int npad)
{
  __shared__ float    sim[16 * 260];
  __shared__ _Float16 phA[256 * 16];   // probs in A-fragment order (max npad=256)
  const int lane = threadIdx.x;
  const int hi  = lane >> 4;
  const int l15 = lane & 15;
  const int h  = blockIdx.y;
  const int sq = blockIdx.z;
  const int qt = blockIdx.x;
  const int q0 = qt * 16;
  const int nqt = npad >> 4, nkk = npad >> 5;
  const size_t hb = (size_t)sq * HEADS + h;

  AF aq0, aq1;
  {
    const _Float16* qp = qsw + (hb * nqt + qt) * 2 * 512 + lane * 16;
    *(uint4*)&aq0.e[0] = *(const uint4*)(qp);
    *(uint4*)&aq0.e[8] = *(const uint4*)(qp + 8);
    *(uint4*)&aq1.e[0] = *(const uint4*)(qp + 512);
    *(uint4*)&aq1.e[8] = *(const uint4*)(qp + 520);
  }

  // sim = (q.k)*8 (+bias), masked beyond n
  for (int jt = 0; jt < nqt; ++jt) {
    AF bk0, bk1;
    const _Float16* kp = ksw + (hb * nqt + jt) * 2 * 512 + lane * 16;
    *(uint4*)&bk0.e[0] = *(const uint4*)(kp);
    *(uint4*)&bk0.e[8] = *(const uint4*)(kp + 8);
    *(uint4*)&bk1.e[0] = *(const uint4*)(kp + 512);
    *(uint4*)&bk1.e[8] = *(const uint4*)(kp + 520);
    CF acc;
    #pragma unroll
    for (int r = 0; r < 8; ++r) acc.e[r] = 0.f;
    acc.v = wmma_f16(aq0.v, bk0.v, acc.v);
    acc.v = wmma_f16(aq1.v, bk1.v, acc.v);
    int j = jt * 16 + l15;
    #pragma unroll
    for (int r = 0; r < 8; ++r) {
      int m = r + hi * 8;
      int i = q0 + m;
      float v = acc.e[r];
      if (btab && i < n && j < n) {
        int di = (i >> 4) - (j >> 4);
        int dj = (i & 15) - (j & 15);
        v += btab[((di + 15) * 31 + (dj + 15)) * HEADS + h];
      }
      if (j >= n || i >= n) v = -1e30f;
      sim[m * 260 + j] = v;
    }
  }
  __syncthreads();

  // softmax per query row; write probs in A-fragment order
  for (int r = 0; r < 16; ++r) {
    float mx = -3e38f;
    for (int j = lane; j < npad; j += 32) mx = fmaxf(mx, sim[r * 260 + j]);
    for (int o = 16; o > 0; o >>= 1) mx = fmaxf(mx, __shfl_xor(mx, o));
    float sum = 0.f;
    for (int j = lane; j < npad; j += 32) {
      float e = expf(sim[r * 260 + j] - mx);
      sim[r * 260 + j] = e;
      sum += e;
    }
    for (int o = 16; o > 0; o >>= 1) sum += __shfl_xor(sum, o);
    float inv = 1.f / fmaxf(sum, 1e-20f);
    for (int j = lane; j < npad; j += 32) {
      int kk = j >> 5, kl = j & 31;
      int lA = (r & 15) + (((kl >> 3) & 1) << 4);
      int eA = ((kl >> 4) << 3) + (kl & 7);
      phA[(kk * 32 + lA) * 16 + eA] = (_Float16)(sim[r * 260 + j] * inv);
    }
  }
  __syncthreads();

  // O = P V via WMMA
  #pragma unroll
  for (int dt = 0; dt < 4; ++dt) {
    CF o;
    #pragma unroll
    for (int r = 0; r < 8; ++r) o.e[r] = 0.f;
    for (int kk = 0; kk < nkk; ++kk) {
      AF ap, bv;
      const _Float16* pp = &phA[(kk * 32 + lane) * 16];
      *(uint4*)&ap.e[0] = *(const uint4*)pp;
      *(uint4*)&ap.e[8] = *(const uint4*)(pp + 8);
      const _Float16* vp = vsw + ((hb * nkk + kk) * 4 + dt) * 512 + lane * 16;
      *(uint4*)&bv.e[0] = *(const uint4*)vp;
      *(uint4*)&bv.e[8] = *(const uint4*)(vp + 8);
      o.v = wmma_f16(ap.v, bv.v, o.v);
    }
    #pragma unroll
    for (int r = 0; r < 8; ++r) {
      int m = r + hi * 8;
      int i = q0 + m;
      if (i < n) out[((size_t)sq * n + i) * DIMV + h * DH + dt * 16 + l15] = o.e[r];
    }
  }
}

// ---------------- LayerNorm: one wave per row ----------------
__global__ __launch_bounds__(32)
void k_layernorm(const float* __restrict__ src, float* __restrict__ dst,
                 const float* __restrict__ g, const float* __restrict__ b, int D)
{
  const int row = blockIdx.x;
  const int lane = threadIdx.x;
  const float* x = src + (size_t)row * D;
  float s1 = 0.f, s2 = 0.f;
  for (int c = lane; c < D; c += 32) { float v = x[c]; s1 += v; s2 += v * v; }
  for (int o = 16; o > 0; o >>= 1) { s1 += __shfl_xor(s1, o); s2 += __shfl_xor(s2, o); }
  float mean = s1 / D;
  float var  = s2 / D - mean * mean;
  float rs   = rsqrtf(var + 1e-5f);
  float* y = dst + (size_t)row * D;
  for (int c = lane; c < D; c += 32) {
    float v = (x[c] - mean) * rs * g[c];
    if (b) v += b[c];
    y[c] = v;
  }
}

// ---------------- q/k l2-norm + scale, emit fragment-blocked f16 q/k/v ----------------
__global__ __launch_bounds__(32)
void k_qkvnorm(const float* __restrict__ q, const float* __restrict__ kv,
               const float* __restrict__ qs, const float* __restrict__ ks,
               _Float16* __restrict__ qsw, _Float16* __restrict__ ksw,
               _Float16* __restrict__ vsw, int n, int npad)
{
  const int h  = blockIdx.x;
  const int rp = blockIdx.y;          // padded row: seq*npad + ti
  const int seq = rp / npad;
  const int ti  = rp % npad;
  const int lane = threadIdx.x;
  const int nqt = npad >> 4, nkk = npad >> 5;
  const size_t hb = (size_t)seq * HEADS + h;

  auto qidx = [&](int t, int d) -> size_t {
    int c = d >> 5, kl = d & 31;
    int lA = (t & 15) + (((kl >> 3) & 1) << 4);
    int eA = ((kl >> 4) << 3) + (kl & 7);
    return ((hb * nqt + (t >> 4)) * 2 + c) * 512 + lA * 16 + eA;
  };
  auto kidx = [&](int t, int d) -> size_t {
    int c = d >> 5, kl = d & 31;
    int lB = (((kl >> 3) & 1) << 4) + (t & 15);
    int eB = ((kl >> 4) << 3) + (kl & 7);
    return ((hb * nqt + (t >> 4)) * 2 + c) * 512 + lB * 16 + eB;
  };
  auto vidx = [&](int t, int d) -> size_t {
    int lB = (((t >> 3) & 1) << 4) + (d & 15);
    int eB = (((t >> 4) & 1) << 3) + (t & 7);
    return ((hb * nkk + (t >> 5)) * 4 + (d >> 4)) * 512 + lB * 16 + eB;
  };

  const int d0 = lane, d1 = lane + 32;
  if (ti >= n) {   // padded token slot: must be zero (esp. V: prob 0 * 0 = 0)
    qsw[qidx(ti, d0)] = (_Float16)0.f; qsw[qidx(ti, d1)] = (_Float16)0.f;
    ksw[kidx(ti, d0)] = (_Float16)0.f; ksw[kidx(ti, d1)] = (_Float16)0.f;
    vsw[vidx(ti, d0)] = (_Float16)0.f; vsw[vidx(ti, d1)] = (_Float16)0.f;
    return;
  }
  const size_t row = (size_t)seq * n + ti;
  const size_t qoff = row * DIMV + h * DH;
  const size_t koff = row * (2 * DIMV) + h * DH;
  const size_t voff = koff + DIMV;
  float q0 = q[qoff + d0],  q1 = q[qoff + d1];
  float k0 = kv[koff + d0], k1 = kv[koff + d1];
  float v0 = kv[voff + d0], v1 = kv[voff + d1];
  float sq = q0 * q0 + q1 * q1, sk = k0 * k0 + k1 * k1;
  for (int o = 16; o > 0; o >>= 1) { sq += __shfl_xor(sq, o); sk += __shfl_xor(sk, o); }
  float iq = 8.0f / fmaxf(sqrtf(sq), 1e-12f);   // SCALE=8 folded into q
  float ik = 1.0f / fmaxf(sqrtf(sk), 1e-12f);
  qsw[qidx(ti, d0)] = (_Float16)(q0 * iq * qs[d0]);
  qsw[qidx(ti, d1)] = (_Float16)(q1 * iq * qs[d1]);
  ksw[kidx(ti, d0)] = (_Float16)(k0 * ik * ks[d0]);
  ksw[kidx(ti, d1)] = (_Float16)(k1 * ik * ks[d1]);
  vsw[vidx(ti, d0)] = (_Float16)v0;
  vsw[vidx(ti, d1)] = (_Float16)v1;
}

// ---------------- PEG: causal depthwise conv3d + bias + residual ----------------
__global__ void k_peg(const float* __restrict__ x, const float* __restrict__ w,
                      const float* __restrict__ pb, float* __restrict__ y)
{
  int idx = blockIdx.x * blockDim.x + threadIdx.x;
  if (idx >= XSZ) return;
  int d = idx & (DIMV - 1);
  int r = idx >> 9;
  int wi = r & 15;  r >>= 4;
  int hh0 = r & 15; r >>= 4;
  int t  = r % TFR;
  int bi = r / TFR;
  float acc = pb[d];
  const float* wd = w + d * 27;
  for (int kt = 0; kt < 3; ++kt) {
    int tt = t - 2 + kt; if (tt < 0) continue;        // temporal pad (2,0)
    for (int kh2 = 0; kh2 < 3; ++kh2) {
      int hh = hh0 - 1 + kh2; if (hh < 0 || hh > 15) continue;
      for (int kw = 0; kw < 3; ++kw) {
        int ww = wi - 1 + kw; if (ww < 0 || ww > 15) continue;
        size_t xi = ((((size_t)bi * TFR + tt) * 16 + hh) * 16 + ww) * DIMV + d;
        acc += x[xi] * wd[kt * 9 + kh2 * 3 + kw];
      }
    }
  }
  y[idx] = acc + x[idx];
}

// ---------------- GEGLU (exact gelu), writes zero-padded K for fw2 ----------------
__global__ void k_geglu(const float* __restrict__ h, float* __restrict__ g, int M)
{
  long long idx = (long long)blockIdx.x * blockDim.x + threadIdx.x;
  long long total = (long long)M * FFIP;
  if (idx >= total) return;
  int i = (int)(idx % FFIP);
  long long m = idx / FFIP;
  float v = 0.f;
  if (i < FFI) {
    float a  = h[m * FF2NP + i];
    float gt = h[m * FF2NP + FFI + i];
    v = a * 0.5f * gt * (1.f + erff(gt * 0.70710678118f));
  }
  g[m * FFIP + i] = v;
}

// ---- fp32 -> f16 weight convert, emitting WMMA B-fragment swizzled layout ----
__global__ void k_cvt(const float* __restrict__ src, _Float16* __restrict__ dst,
                      int K, int N, int Np, int total)
{
  int idx = blockIdx.x * blockDim.x + threadIdx.x;
  if (idx >= total) return;
  int n = idx % Np;
  int k = idx / Np;
  int kt = k >> 5, nt = n >> 4;
  int lB = (((k >> 3) & 1) << 4) + (n & 15);
  int eB = (((k >> 4) & 1) << 3) + (k & 7);
  size_t di = (((size_t)kt * (Np >> 4) + nt) * 32 + lB) * 16 + eB;
  dst[di] = (k < K && n < N) ? (_Float16)src[(size_t)k * N + n] : (_Float16)0.f;
}

// ---------------- patch extraction ----------------
__global__ void k_patch1(const float* __restrict__ vid, float* __restrict__ out)
{
  int idx = blockIdx.x * blockDim.x + threadIdx.x;   // 512 rows x 256 cols
  if (idx >= 512 * 256) return;
  int c = idx & 255, r = idx >> 8;
  int wi = r & 15, hh = (r >> 4) & 15, bi = r >> 8;
  int p1 = c >> 4, p2 = c & 15;
  size_t vi = (size_t)bi * 33 * 65536 + (size_t)(hh * 16 + p1) * 256 + (wi * 16 + p2);
  out[idx] = vid[vi];
}

__global__ void k_patch2(const float* __restrict__ vid, float* __restrict__ out)
{
  int idx = blockIdx.x * blockDim.x + threadIdx.x;   // 8192 rows x 512 cols
  if (idx >= 8192 * 512) return;
  int c = idx & 511, r = idx >> 9;
  int wi = r & 15, hh = (r >> 4) & 15, tt = (r >> 8) & 15, bi = r >> 12;
  int pt = c >> 8, p1 = (c >> 4) & 15, p2 = c & 15;
  int f = 1 + tt * 2 + pt;
  size_t vi = (size_t)bi * 33 * 65536 + (size_t)f * 65536 +
              (size_t)(hh * 16 + p1) * 256 + (wi * 16 + p2);
  out[idx] = vid[vi];
}

// scatter embedded rows into token buffer (b,t,h,w,d)
__global__ void k_scatter(const float* __restrict__ src, float* __restrict__ dst,
                          int rows_per_b, int t_off, int nrows)
{
  int idx = blockIdx.x * blockDim.x + threadIdx.x;
  if (idx >= nrows * DIMV) return;
  int d = idx & 511, r = idx >> 9;
  int bi = r / rows_per_b, rem = r % rows_per_b;
  size_t dr = (size_t)bi * (TFR * 256) + (size_t)t_off * 256 + rem;
  dst[dr * DIMV + d] = src[idx];
}

// ---------------- relpos bias: 961 unique (di,dj) offsets only ----------------
__global__ void k_relpos_l1(const float* __restrict__ w1, const float* __restrict__ b1,
                            float* __restrict__ o)
{
  int idx = blockIdx.x * blockDim.x + threadIdx.x;
  if (idx >= 961 * DIMV) return;
  int nn = idx & 511, p = idx >> 9;
  float di = (float)(p / 31 - 15), dj = (float)(p % 31 - 15);
  float f0 = (di > 0.f ? 1.f : (di < 0.f ? -1.f : 0.f)) * logf(fabsf(di) + 1.f);
  float f1 = (dj > 0.f ? 1.f : (dj < 0.f ? -1.f : 0.f)) * logf(fabsf(dj) + 1.f);
  float v = f0 * w1[nn] + f1 * w1[DIMV + nn] + b1[nn];
  o[idx] = v < 0.f ? 0.1f * v : v;
}

__global__ void k_relpos_l3(const float* __restrict__ x, const float* __restrict__ w3,
                            const float* __restrict__ b3, float* __restrict__ tab)
{
  int idx = blockIdx.x * blockDim.x + threadIdx.x;
  if (idx >= 961 * HEADS) return;
  int h = idx & 7, p = idx >> 3;
  float s = b3[h];
  for (int k = 0; k < DIMV; ++k) s += x[(size_t)p * DIMV + k] * w3[k * HEADS + h];
  tab[idx] = s;
}

// ---------------- layout transposes (b,t,h,w,d) <-> (b,h,w,t,d) ----------------
__global__ void k_tr_fwd(const float* __restrict__ src, float* __restrict__ dst)
{
  int idx = blockIdx.x * blockDim.x + threadIdx.x;
  if (idx >= XSZ) return;
  int d = idx & 511, r = idx >> 9;
  int wi = r & 15, hh = (r >> 4) & 15;
  int bt = r >> 8;
  int t = bt % TFR, bi = bt / TFR;
  size_t dr = (((size_t)bi * 16 + hh) * 16 + wi) * TFR + t;
  dst[dr * DIMV + d] = src[idx];
}

__global__ void k_tr_bwd(const float* __restrict__ src, float* __restrict__ dst)
{
  int idx = blockIdx.x * blockDim.x + threadIdx.x;
  if (idx >= XSZ) return;
  int d = idx & 511, r = idx >> 9;        // src row in (b,h,w,t)
  int t = r % TFR, q2 = r / TFR;
  int wi = q2 & 15, hh = (q2 >> 4) & 15, bi = q2 >> 8;
  size_t dr = ((size_t)bi * TFR + t) * 256 + hh * 16 + wi;
  dst[dr * DIMV + d] = src[idx];
}

// ================= host orchestration =================
static size_t alignup(size_t x) { return (x + 255) & ~(size_t)255; }

extern "C" void kernel_launch(void* const* d_in, const int* in_sizes, int n_in,
                              void* d_out, int out_size, void* d_ws, size_t ws_size,
                              hipStream_t stream)
{
  (void)in_sizes; (void)n_in; (void)out_size; (void)ws_size;
  const float* video    = (const float*)d_in[0];
  const float* p_ln1_g  = (const float*)d_in[1];
  const float* p_ln1_b  = (const float*)d_in[2];
  const float* p_w1     = (const float*)d_in[3];
  const float* p_b1     = (const float*)d_in[4];
  const float* p_ln1o_g = (const float*)d_in[5];
  const float* p_ln1o_b = (const float*)d_in[6];
  const float* p_ln2_g  = (const float*)d_in[7];
  const float* p_ln2_b  = (const float*)d_in[8];
  const float* p_w2     = (const float*)d_in[9];
  const float* p_b2     = (const float*)d_in[10];
  const float* p_ln2o_g = (const float*)d_in[11];
  const float* p_ln2o_b = (const float*)d_in[12];
  const float* rp_w1 = (const float*)d_in[13];
  const float* rp_b1 = (const float*)d_in[14];
  const float* rp_w2 = (const float*)d_in[15];
  const float* rp_b2 = (const float*)d_in[16];
  const float* rp_w3 = (const float*)d_in[17];
  const float* rp_b3 = (const float*)d_in[18];
  const float* SP[13]; for (int i = 0; i < 13; ++i) SP[i] = (const float*)d_in[19 + i];
  const float* TP[13]; for (int i = 0; i < 13; ++i) TP[i] = (const float*)d_in[32 + i];
  // order: 0 peg_w 1 peg_b 2 ag 3 wq 4 wkv 5 qs 6 ks 7 wo 8 fg 9 fb 10 fw1 11 fw2 12 ng

  // ---- workspace carve ----
  char* w = (char*)d_ws;
  auto take = [&](size_t bytes) { char* p = w; w += alignup(bytes); return p; };
  float* x0  = (float*)take((size_t)XSZ * 4);
  float* x1  = (float*)take((size_t)XSZ * 4);
  float* xn  = (float*)take((size_t)XSZ * 4);
  float* q   = (float*)take((size_t)XSZ * 4);
  float* kv  = (float*)take((size_t)XSZ * 8);
  float* ao  = (float*)take((size_t)XSZ * 4);
  float* ffh = (float*)take((size_t)NTOK * FF2NP * 4);
  float* ffg = (float*)take((size_t)NTOK * FFIP * 4);
  _Float16* qsw = (_Float16*)take((size_t)QKVPAD * 2);
  _Float16* ksw = (_Float16*)take((size_t)QKVPAD * 2);
  _Float16* vsw = (_Float16*)take((size_t)QKVPAD * 2);
  _Float16* wqh  = (_Float16*)take((size_t)DIMV * DIMV * 2);
  _Float16* wkvh = (_Float16*)take((size_t)DIMV * 2 * DIMV * 2);
  _Float16* woh  = (_Float16*)take((size_t)DIMV * DIMV * 2);
  _Float16* fw1h = (_Float16*)take((size_t)DIMV * FF2NP * 2);
  _Float16* fw2h = (_Float16*)take((size_t)FFIP * DIMV * 2);
  float* p1b = (float*)take((size_t)512 * 256 * 4);
  float* e1  = (float*)take((size_t)512 * DIMV * 4);
  float* p2b = (float*)take((size_t)8192 * DIMV * 4);
  float* e2  = (float*)take((size_t)8192 * DIMV * 4);
  float* rp1 = (float*)take((size_t)961 * DIMV * 4);
  float* rp2 = (float*)take((size_t)961 * DIMV * 4);
  float* tab = (float*)take((size_t)961 * HEADS * 4);

  hipStream_t st = stream;
  auto LN = [&](const float* s, float* d, const float* g, const float* b, int M, int D) {
    k_layernorm<<<dim3(M), dim3(32), 0, st>>>(s, d, g, b, D);
  };
  auto GEMM = [&](const float* A, const _Float16* B, const float* bias,
                  const float* res, float* C, int M, int N, int K, int act) {
    k_gemm<<<dim3((N + BN - 1) / BN, (M + BM - 1) / BM), dim3(256), 0, st>>>(
        A, B, bias, res, C, M, N, K, act);
  };
  auto CVT = [&](const float* s, _Float16* d, int K, int N, int Kp, int Np) {
    int total = Kp * Np;
    k_cvt<<<dim3((total + 255) / 256), dim3(256), 0, st>>>(s, d, K, N, Np, total);
  };

  // ---- patch embedding ----
  CVT(p_w1, wqh, 256, DIMV, 256, DIMV);
  CVT(p_w2, wkvh, DIMV, DIMV, DIMV, DIMV);
  k_patch1<<<dim3((512 * 256 + 255) / 256), dim3(256), 0, st>>>(video, p1b);
  LN(p1b, p1b, p_ln1_g, p_ln1_b, 512, 256);
  GEMM(p1b, wqh, p_b1, nullptr, e1, 512, DIMV, 256, 0);
  LN(e1, e1, p_ln1o_g, p_ln1o_b, 512, DIMV);
  k_scatter<<<dim3((512 * DIMV + 255) / 256), dim3(256), 0, st>>>(e1, x0, 256, 0, 512);
  k_patch2<<<dim3((8192 * 512 + 255) / 256), dim3(256), 0, st>>>(video, p2b);
  LN(p2b, p2b, p_ln2_g, p_ln2_b, 8192, DIMV);
  GEMM(p2b, wkvh, p_b2, nullptr, e2, 8192, DIMV, DIMV, 0);
  LN(e2, e2, p_ln2o_g, p_ln2o_b, 8192, DIMV);
  k_scatter<<<dim3((8192 * DIMV + 255) / 256), dim3(256), 0, st>>>(e2, x0, 4096, 1, 8192);

  // ---- relpos bias table on 31x31 unique offsets ----
  CVT(rp_w2, wqh, DIMV, DIMV, DIMV, DIMV);
  k_relpos_l1<<<dim3((961 * DIMV + 255) / 256), dim3(256), 0, st>>>(rp_w1, rp_b1, rp1);
  GEMM(rp1, wqh, rp_b2, nullptr, rp2, 961, DIMV, DIMV, 1);
  k_relpos_l3<<<dim3((961 * HEADS + 255) / 256), dim3(256), 0, st>>>(rp2, rp_w3, rp_b3, tab);

  // ---- transformer stages ----
  float* X = x0;
  float* Y = x1;
  auto swapXY = [&]() { float* t = X; X = Y; Y = t; };

  auto run_stage = [&](const float* const* P, const float* bias_tab,
                       int n, int npad, int nseq) {
    for (int i = 0; i < 4; ++i) {
      CVT(P[3]  + (size_t)i * DIMV * DIMV,     wqh,  DIMV, DIMV,   DIMV, DIMV);
      CVT(P[4]  + (size_t)i * DIMV * 2 * DIMV, wkvh, DIMV, 2*DIMV, DIMV, 2*DIMV);
      CVT(P[7]  + (size_t)i * DIMV * DIMV,     woh,  DIMV, DIMV,   DIMV, DIMV);
      CVT(P[10] + (size_t)i * DIMV * FF2N,     fw1h, DIMV, FF2N,   DIMV, FF2NP);
      CVT(P[11] + (size_t)i * FFI * DIMV,      fw2h, FFI,  DIMV,   FFIP, DIMV);
      // PEG + residual
      k_peg<<<dim3((XSZ + 255) / 256), dim3(256), 0, st>>>(
          X, P[0] + (size_t)i * DIMV * 27, P[1] + i * DIMV, Y);
      swapXY();
      // attention block
      LN(X, xn, P[2] + i * DIMV, nullptr, NTOK, DIMV);
      GEMM(xn, wqh,  nullptr, nullptr, q,  NTOK, DIMV,     DIMV, 0);
      GEMM(xn, wkvh, nullptr, nullptr, kv, NTOK, 2 * DIMV, DIMV, 0);
      k_qkvnorm<<<dim3(HEADS, nseq * npad), dim3(32), 0, st>>>(
          q, kv, P[5] + i * DH, P[6] + i * DH, qsw, ksw, vsw, n, npad);
      k_attn<<<dim3(npad / 16, HEADS, nseq), dim3(32), 0, st>>>(
          qsw, ksw, vsw, bias_tab, ao, n, npad);
      GEMM(ao, woh, nullptr, X, Y, NTOK, DIMV, DIMV, 0);
      swapXY();
      // GEGLU FF block
      LN(X, xn, P[8] + i * DIMV, P[9] + i * DIMV, NTOK, DIMV);
      GEMM(xn, fw1h, nullptr, nullptr, ffh, NTOK, FF2NP, DIMV, 0);
      k_geglu<<<dim3((NTOK * FFIP + 255) / 256), dim3(256), 0, st>>>(ffh, ffg, NTOK);
      GEMM(ffg, fw2h, nullptr, X, Y, NTOK, DIMV, FFIP, 0);
      swapXY();
    }
    LN(X, xn, P[12], nullptr, NTOK, DIMV);   // final stage LN -> xn
  };

  // spatial: 34 sequences of 256 tokens, relpos bias
  run_stage(SP, tab, 256, 256, 34);
  // (b,t,h,w,d) -> (b,h,w,t,d)
  k_tr_fwd<<<dim3((XSZ + 255) / 256), dim3(256), 0, st>>>(xn, Y);
  swapXY();
  // temporal: 512 sequences of 17 tokens (padded to 32), no bias
  run_stage(TP, nullptr, 17, 32, 512);
  // (b,h,w,t,d) -> (b,t,h,w,d) into d_out
  k_tr_bwd<<<dim3((XSZ + 255) / 256), dim3(256), 0, st>>>(xn, (float*)d_out);
}
